// Encoder_46153718563052
// MI455X (gfx1250) — compile-verified, matrix operand order checked by reading, not run on hardware
//
#include <hip/hip_runtime.h>
#include <cstdint>

#define BB 4
#define TT 2048
#define DD 1024
#define HH 16
#define HDIM 64
#define DFFN 4096
#define MROWS (BB*TT)

typedef __attribute__((ext_vector_type(16))) __bf16 v16bf;
typedef __attribute__((ext_vector_type(8)))  float  v8f;

union Frag16 { v16bf v; unsigned int u[8]; unsigned short s[16]; };
union Pack8  { uint4 q; unsigned short s[8]; };

__device__ __forceinline__ unsigned short f2bf(float f) {
  unsigned int x = __float_as_uint(f);
  x += 0x7fffu + ((x >> 16) & 1u);          // round-to-nearest-even
  return (unsigned short)(x >> 16);
}

// LDS byte offset of a shared-memory pointer: generic LDS addresses carry the
// workgroup-relative offset in addr[31:0] (aperture mapping, ISA 10.2).
__device__ __forceinline__ unsigned lds_off(const void* p) {
  return (unsigned)(unsigned long long)(uintptr_t)p;
}

// Async global->LDS 128-bit copy (GV mode, ASYNCcnt-tracked).
__device__ __forceinline__ void async_b128(unsigned lds, const unsigned short* g) {
  asm volatile("global_load_async_to_lds_b128 %0, %1, off"
               :: "v"(lds), "v"(g) : "memory");
}
__device__ __forceinline__ void wait_async_le(int n) {
  if (n == 0)      asm volatile("s_wait_asynccnt 0x0" ::: "memory");
  else if (n == 2) asm volatile("s_wait_asynccnt 0x2" ::: "memory");
  else             asm volatile("s_wait_asynccnt 0x8" ::: "memory");
}

// Load the 16-bit A/B fragment pattern from a 32-element row:
// lane kbase=(lane>=16)?8:0; VGPR j<4 -> K=kbase+2j,2j+1 ; j>=4 -> +16.
// Pairs are contiguous, so this folds into wide ds loads.
__device__ __forceinline__ void load_pair_frag(const unsigned short* row, int kb, Frag16& f) {
#pragma unroll
  for (int j = 0; j < 4; ++j) f.u[j] = *(const unsigned int*)(row + kb + 2 * j);
#pragma unroll
  for (int j = 4; j < 8; ++j) f.u[j] = *(const unsigned int*)(row + 8 + kb + 2 * j);
}

__device__ __forceinline__ v8f zero8() {
  v8f z = {0.f,0.f,0.f,0.f,0.f,0.f,0.f,0.f};
  return z;
}

// ---------------- transpose + cast f32 -> bf16: dst[n][k] = src[k][n] ----------------
__global__ __launch_bounds__(256) void enc_castT(const float* __restrict__ src,
                                                 unsigned short* __restrict__ dst,
                                                 int K, int N) {
  __shared__ unsigned short tile[32][33];
  const int kt = blockIdx.y * 32, nt = blockIdx.x * 32;
  const int tx = threadIdx.x & 31, ty = threadIdx.x >> 5;
#pragma unroll
  for (int i = ty; i < 32; i += 8)
    tile[i][tx] = f2bf(src[(size_t)(kt + i) * N + (nt + tx)]);
  __syncthreads();
#pragma unroll
  for (int i = ty; i < 32; i += 8)
    dst[(size_t)(nt + i) * K + (kt + tx)] = tile[tx][i];
}

// ---------------- LayerNorm (one block per row of D=1024, 4 elems/thread) ----------------
__global__ __launch_bounds__(256) void enc_ln(const float* __restrict__ x,
                                              const float* __restrict__ g,
                                              const float* __restrict__ bta,
                                              float* __restrict__ outF,          // may be null
                                              unsigned short* __restrict__ outH) {
  const int row = blockIdx.x;
  const int i4 = threadIdx.x * 4;
  const float4 xv = *(const float4*)(x + (size_t)row * DD + i4);
  float s  = xv.x + xv.y + xv.z + xv.w;
  float s2 = xv.x * xv.x + xv.y * xv.y + xv.z * xv.z + xv.w * xv.w;
#pragma unroll
  for (int o = 1; o < 32; o <<= 1) { s += __shfl_xor(s, o, 32); s2 += __shfl_xor(s2, o, 32); }
  __shared__ float shS[8], shS2[8];
  __shared__ float shMean, shRstd;
  const int wave = threadIdx.x >> 5, lane = threadIdx.x & 31;
  if (lane == 0) { shS[wave] = s; shS2[wave] = s2; }
  __syncthreads();
  if (threadIdx.x == 0) {
    float ts = 0.f, ts2 = 0.f;
#pragma unroll
    for (int i = 0; i < 8; ++i) { ts += shS[i]; ts2 += shS2[i]; }
    float mean = ts / (float)DD;
    float var  = ts2 / (float)DD - mean * mean;
    shMean = mean;
    shRstd = rsqrtf(var + 1e-5f);
  }
  __syncthreads();
  const float mean = shMean, rstd = shRstd;
  const float4 gv = *(const float4*)(g + i4);
  const float4 bv = *(const float4*)(bta + i4);
  float y0 = (xv.x - mean) * rstd * gv.x + bv.x;
  float y1 = (xv.y - mean) * rstd * gv.y + bv.y;
  float y2 = (xv.z - mean) * rstd * gv.z + bv.z;
  float y3 = (xv.w - mean) * rstd * gv.w + bv.w;
  if (outF) {
    float4 yv = make_float4(y0, y1, y2, y3);
    *(float4*)(outF + (size_t)row * DD + i4) = yv;
  }
  uint2 pk;
  pk.x = (unsigned)f2bf(y0) | ((unsigned)f2bf(y1) << 16);
  pk.y = (unsigned)f2bf(y2) | ((unsigned)f2bf(y3) << 16);
  *(uint2*)(outH + (size_t)row * DD + i4) = pk;
}

// ---------------- Tiled bf16 WMMA GEMM: C = A[MxK] * B, B given transposed BT[NxK] ----------------
// 128x128 block tile, 8 waves (4m x 2n), each wave 32x64; K-step 64; async double-buffered LDS.
// mode 0: outH = bf16(C+bias)
// mode 1: outF = resid + C + bias          (f32)
// mode 2: outH = bf16(gelu_exact(C+bias))
__global__ __launch_bounds__(256) void enc_gemm_bf16(
    const unsigned short* __restrict__ A, const unsigned short* __restrict__ BT,
    const float* __restrict__ bias, const float* __restrict__ resid,
    float* __restrict__ outF, unsigned short* __restrict__ outH,
    int Mdim, int Ndim, int Kdim, int mode) {
  __shared__ unsigned short As[2][128][72];   // rows 144B (16B aligned)
  __shared__ unsigned short Bs[2][128][72];
  const int tid  = threadIdx.x;
  const int lane = tid & 31, wave = tid >> 5;
  const int wm = wave & 3, wn = wave >> 2;    // 4 x 2 -> 128 x 128 tile
  const int mblk = blockIdx.y * 128;
  const int nblk = blockIdx.x * 128;
  const int l15 = lane & 15;
  const int kb  = (lane >> 4) << 3;

  v8f c[2][4];
#pragma unroll
  for (int hm = 0; hm < 2; ++hm)
#pragma unroll
    for (int nt = 0; nt < 4; ++nt) c[hm][nt] = zero8();

  const int sRow = tid >> 1;            // 0..127
  const int sK   = (tid & 1) * 32;      // 0 or 32

  auto stage = [&](int k0, int buf) {
    const unsigned short* ga = A  + (size_t)(mblk + sRow) * Kdim + k0 + sK;
    const unsigned short* gb = BT + (size_t)(nblk + sRow) * Kdim + k0 + sK;
    unsigned la = lds_off(&As[buf][sRow][sK]);
    unsigned lb = lds_off(&Bs[buf][sRow][sK]);
#pragma unroll
    for (int j = 0; j < 4; ++j) {
      async_b128(la + j * 16, ga + j * 8);
      async_b128(lb + j * 16, gb + j * 8);
    }
  };

  const int nSteps = Kdim >> 6;
  stage(0, 0);
  for (int s = 0; s < nSteps; ++s) {
    const int buf = s & 1;
    const bool hasNext = (s + 1) < nSteps;
    if (hasNext) stage((s + 1) << 6, buf ^ 1);
    wait_async_le(hasNext ? 8 : 0);     // in-order retirement -> current tile complete
    __syncthreads();
#pragma unroll
    for (int ks = 0; ks < 2; ++ks) {
      // batch all fragment loads, then burst the 8 WMMAs
      Frag16 a0, a1, bf[4];
      load_pair_frag(&As[buf][wm * 32 + l15][ks * 32],      kb, a0);
      load_pair_frag(&As[buf][wm * 32 + 16 + l15][ks * 32], kb, a1);
#pragma unroll
      for (int nt = 0; nt < 4; ++nt)
        load_pair_frag(&Bs[buf][wn * 64 + nt * 16 + l15][ks * 32], kb, bf[nt]);
#pragma unroll
      for (int nt = 0; nt < 4; ++nt) {
        c[0][nt] = __builtin_amdgcn_wmma_f32_16x16x32_bf16(false, a0.v, false, bf[nt].v,
                                                           (short)0, c[0][nt], false, false);
        c[1][nt] = __builtin_amdgcn_wmma_f32_16x16x32_bf16(false, a1.v, false, bf[nt].v,
                                                           (short)0, c[1][nt], false, false);
      }
    }
    __syncthreads();
  }

  // epilogue: C layout -> row = r + 8*(lane>=16), col = lane&15 within a 16x16 tile
  const int rowOff = (lane >> 4) << 3;
#pragma unroll
  for (int hm = 0; hm < 2; ++hm) {
    const int rowBase = mblk + wm * 32 + hm * 16 + rowOff;
#pragma unroll
    for (int nt = 0; nt < 4; ++nt) {
      const int col = nblk + wn * 64 + nt * 16 + l15;
      const float bv = bias ? bias[col] : 0.0f;
#pragma unroll
      for (int r = 0; r < 8; ++r) {
        const size_t o = (size_t)(rowBase + r) * Ndim + col;
        float v = c[hm][nt][r] + bv;
        if (mode == 1) {
          outF[o] = resid[o] + v;
        } else if (mode == 2) {
          float gl = 0.5f * v * (1.0f + erff(v * 0.70710678118654752f));
          outH[o] = f2bf(gl);
        } else {
          outH[o] = f2bf(v);
        }
      }
    }
  }
}

// ---------------- Flash attention (bf16 WMMA QK^T and PV, f32 online softmax) ----------------
// grid = (T/64, H, B), 128 threads (4 waves); each wave owns a 16-row Q strip.
// K tiles async double-buffered; V tiles staged transposed (Vt[hd][key]) for wide fragment loads.
__global__ __launch_bounds__(128) void enc_attn(
    const unsigned short* __restrict__ Qh, const unsigned short* __restrict__ Kh,
    const unsigned short* __restrict__ Vh, const float* __restrict__ mask,
    unsigned short* __restrict__ Oh) {
  __shared__ unsigned short Kt[2][32][72];     // 32 keys x 64 hd
  __shared__ unsigned short Vt[2][64][40];     // 64 hd x 32 keys (transposed)
  __shared__ unsigned short Pt[4][16][34];     // per-wave P tile (16 q x 32 keys)
  const int tid  = threadIdx.x;
  const int lane = tid & 31, wave = tid >> 5;
  const int qblk = blockIdx.x * 64;
  const int h = blockIdx.y;
  const int b = blockIdx.z;
  const int l15 = lane & 15;
  const int kb  = (lane >> 4) << 3;
  const int rowOff = (lane >> 4) << 3;
  const float scale = 0.125f;                  // HD^-0.5

  // Q strip fragments, kept in registers for the whole K loop
  const int qr = qblk + wave * 16 + l15;
  Frag16 aq0, aq1;
  {
    const unsigned short* qrow = Qh + (size_t)(b * TT + qr) * DD + h * HDIM;
    load_pair_frag(qrow,      kb, aq0);        // hd 0..31
    load_pair_frag(qrow + 32, kb, aq1);        // hd 32..63
  }

  v8f acc[4];
#pragma unroll
  for (int i = 0; i < 4; ++i) acc[i] = zero8();
  float mrow[8], lrow[8];
#pragma unroll
  for (int r = 0; r < 8; ++r) { mrow[r] = -1e30f; lrow[r] = 0.f; }

  const int sIdx = tid * 16;                   // 32*64 elems / 128 threads
  const int kr = sIdx >> 6, kc = sIdx & 63;

  auto stageK = [&](int kk0, int buf) {
    const unsigned short* g = Kh + (size_t)(b * TT + kk0 + kr) * DD + h * HDIM + kc;
    unsigned l = lds_off(&Kt[buf][kr][kc]);
    async_b128(l, g);
    async_b128(l + 16, g + 8);
  };
  auto stageV = [&](int kk0, int buf) {        // transpose through registers
    const unsigned short* g = Vh + (size_t)(b * TT + kk0 + kr) * DD + h * HDIM + kc;
    Pack8 p0, p1;
    p0.q = *(const uint4*)g;
    p1.q = *(const uint4*)(g + 8);
#pragma unroll
    for (int j = 0; j < 8; ++j) {
      Vt[buf][kc + j][kr]     = p0.s[j];
      Vt[buf][kc + 8 + j][kr] = p1.s[j];
    }
  };

  const int nSteps = TT / 32;
  stageK(0, 0);
  stageV(0, 0);
  for (int s = 0; s < nSteps; ++s) {
    const int buf = s & 1;
    const bool hasNext = (s + 1) < nSteps;
    const int kk0 = s * 32;
    if (hasNext) { stageK(kk0 + 32, buf ^ 1); stageV(kk0 + 32, buf ^ 1); }
    wait_async_le(hasNext ? 2 : 0);
    __syncthreads();

    // --- scores: two 16-key tiles; batch K-fragment loads then burst 4 WMMAs ---
    Frag16 bk[2][2];
#pragma unroll
    for (int half = 0; half < 2; ++half) {
      const unsigned short* krow = &Kt[buf][half * 16 + l15][0]; // B-frag: n = key, k = hd
      load_pair_frag(krow,      kb, bk[half][0]);
      load_pair_frag(krow + 32, kb, bk[half][1]);
    }
    v8f sv0 = zero8(), sv1 = zero8();
    sv0 = __builtin_amdgcn_wmma_f32_16x16x32_bf16(false, aq0.v, false, bk[0][0].v, (short)0, sv0, false, false);
    sv0 = __builtin_amdgcn_wmma_f32_16x16x32_bf16(false, aq1.v, false, bk[0][1].v, (short)0, sv0, false, false);
    sv1 = __builtin_amdgcn_wmma_f32_16x16x32_bf16(false, aq0.v, false, bk[1][0].v, (short)0, sv1, false, false);
    sv1 = __builtin_amdgcn_wmma_f32_16x16x32_bf16(false, aq1.v, false, bk[1][1].v, (short)0, sv1, false, false);

    float sc[2][8];
#pragma unroll
    for (int r = 0; r < 8; ++r) {
      const int q = qblk + wave * 16 + r + rowOff;
      const float* mrow_p = mask + (size_t)b * TT * TT + (size_t)q * TT + kk0;
      sc[0][r] = sv0[r] * scale + mrow_p[l15];
      sc[1][r] = sv1[r] * scale + mrow_p[16 + l15];
    }

    // --- online softmax (row reductions across the 16-lane half-group) ---
    float pv[2][8];
#pragma unroll
    for (int r = 0; r < 8; ++r) {
      float mx = fmaxf(sc[0][r], sc[1][r]);
#pragma unroll
      for (int o = 1; o < 16; o <<= 1) mx = fmaxf(mx, __shfl_xor(mx, o, 32));
      float mnew = fmaxf(mrow[r], mx);
      float fr = __expf(mrow[r] - mnew);
      float p0 = __expf(sc[0][r] - mnew);
      float p1 = __expf(sc[1][r] - mnew);
      pv[0][r] = p0; pv[1][r] = p1;
      float rs = p0 + p1;
#pragma unroll
      for (int o = 1; o < 16; o <<= 1) rs += __shfl_xor(rs, o, 32);
      lrow[r] = lrow[r] * fr + rs;
      mrow[r] = mnew;
#pragma unroll
      for (int nt = 0; nt < 4; ++nt) acc[nt][r] *= fr;
    }

    // --- P: C-layout -> A-layout via wave-private LDS round trip ---
#pragma unroll
    for (int r = 0; r < 8; ++r) {
      Pt[wave][r + rowOff][l15]      = f2bf(pv[0][r]);
      Pt[wave][r + rowOff][16 + l15] = f2bf(pv[1][r]);
    }
    Frag16 ap;
    load_pair_frag(&Pt[wave][l15][0], kb, ap);

    // --- P @ V: batch all 4 V-fragments then burst 4 WMMAs ---
    Frag16 bv[4];
#pragma unroll
    for (int nt = 0; nt < 4; ++nt)
      load_pair_frag(&Vt[buf][nt * 16 + l15][0], kb, bv[nt]);
#pragma unroll
    for (int nt = 0; nt < 4; ++nt)
      acc[nt] = __builtin_amdgcn_wmma_f32_16x16x32_bf16(false, ap.v, false, bv[nt].v,
                                                        (short)0, acc[nt], false, false);
    __syncthreads();
  }

  // normalize and write attention output (B,T,H,HD) == (B,T,D) row-major, bf16
#pragma unroll
  for (int r = 0; r < 8; ++r) {
    const int q = qblk + wave * 16 + r + rowOff;
    float inv = (lrow[r] > 0.f) ? 1.0f / lrow[r] : 0.f;
#pragma unroll
    for (int nt = 0; nt < 4; ++nt) {
      Oh[(size_t)(b * TT + q) * DD + h * HDIM + nt * 16 + l15] = f2bf(acc[nt][r] * inv);
    }
  }
}

// ---------------- host-side orchestration ----------------
extern "C" void kernel_launch(void* const* d_in, const int* in_sizes, int n_in,
                              void* d_out, int out_size, void* d_ws, size_t ws_size,
                              hipStream_t stream) {
  const float* x    = (const float*)d_in[0];
  const float* mask = (const float*)d_in[1];
  const float* ln1g = (const float*)d_in[2];
  const float* ln1b = (const float*)d_in[3];
  const float* wq   = (const float*)d_in[4];
  const float* wk   = (const float*)d_in[5];
  const float* wv   = (const float*)d_in[6];
  const float* wo   = (const float*)d_in[7];
  const float* bo   = (const float*)d_in[8];
  const float* ln2g = (const float*)d_in[9];
  const float* ln2b = (const float*)d_in[10];
  const float* w1   = (const float*)d_in[11];
  const float* b1   = (const float*)d_in[12];
  const float* w2   = (const float*)d_in[13];
  const float* b2   = (const float*)d_in[14];
  float* out = (float*)d_out;

  char* cur = (char*)d_ws;
  auto alloc = [&](size_t bytes) -> void* {
    void* p = (void*)cur;
    cur += (bytes + 255) & ~(size_t)255;
    return p;
  };

  float*          zf    = (float*)         alloc((size_t)MROWS * DD * 4);   // LN1 out f32 (residual)
  unsigned short* zh    = (unsigned short*)alloc((size_t)MROWS * DD * 2);   // LN1 out bf16
  unsigned short* wqh   = (unsigned short*)alloc((size_t)DD * DD * 2);      // all weights transposed [N][K]
  unsigned short* wkh   = (unsigned short*)alloc((size_t)DD * DD * 2);
  unsigned short* wvh   = (unsigned short*)alloc((size_t)DD * DD * 2);
  unsigned short* woh   = (unsigned short*)alloc((size_t)DD * DD * 2);
  unsigned short* w1h   = (unsigned short*)alloc((size_t)DD * DFFN * 2);
  unsigned short* w2h   = (unsigned short*)alloc((size_t)DFFN * DD * 2);
  unsigned short* Qh    = (unsigned short*)alloc((size_t)MROWS * DD * 2);
  unsigned short* Kh    = (unsigned short*)alloc((size_t)MROWS * DD * 2);
  unsigned short* Vh    = (unsigned short*)alloc((size_t)MROWS * DD * 2);
  unsigned short* attnh = (unsigned short*)alloc((size_t)MROWS * DD * 2);
  float*          z1f   = (float*)         alloc((size_t)MROWS * DD * 4);   // post-attn residual f32
  unsigned short* z2h   = (unsigned short*)alloc((size_t)MROWS * DD * 2);   // LN2 out bf16
  unsigned short* hh    = (unsigned short*)alloc((size_t)MROWS * DFFN * 2); // FFN hidden bf16

  // 1) transpose-cast weights to bf16 [N][K]
  {
    dim3 gDD(DD / 32, DD / 32);
    enc_castT<<<gDD, 256, 0, stream>>>(wq, wqh, DD, DD);
    enc_castT<<<gDD, 256, 0, stream>>>(wk, wkh, DD, DD);
    enc_castT<<<gDD, 256, 0, stream>>>(wv, wvh, DD, DD);
    enc_castT<<<gDD, 256, 0, stream>>>(wo, woh, DD, DD);
    dim3 g1(DFFN / 32, DD / 32);   // w1: K=DD, N=DFFN
    enc_castT<<<g1, 256, 0, stream>>>(w1, w1h, DD, DFFN);
    dim3 g2(DD / 32, DFFN / 32);   // w2: K=DFFN, N=DD
    enc_castT<<<g2, 256, 0, stream>>>(w2, w2h, DFFN, DD);
  }

  // 2) LN1
  enc_ln<<<MROWS, 256, 0, stream>>>(x, ln1g, ln1b, zf, zh);

  // 3) Q/K/V projections (no bias)
  dim3 gD(DD / 128, MROWS / 128);
  enc_gemm_bf16<<<gD, 256, 0, stream>>>(zh, wqh, nullptr, nullptr, nullptr, Qh, MROWS, DD, DD, 0);
  enc_gemm_bf16<<<gD, 256, 0, stream>>>(zh, wkh, nullptr, nullptr, nullptr, Kh, MROWS, DD, DD, 0);
  enc_gemm_bf16<<<gD, 256, 0, stream>>>(zh, wvh, nullptr, nullptr, nullptr, Vh, MROWS, DD, DD, 0);

  // 4) flash attention
  dim3 gA(TT / 64, HH, BB);
  enc_attn<<<gA, 128, 0, stream>>>(Qh, Kh, Vh, mask, attnh);

  // 5) output projection + bo + residual (z)
  enc_gemm_bf16<<<gD, 256, 0, stream>>>(attnh, woh, bo, zf, z1f, nullptr, MROWS, DD, DD, 1);

  // 6) LN2
  enc_ln<<<MROWS, 256, 0, stream>>>(z1f, ln2g, ln2b, nullptr, z2h);

  // 7) FFN up + exact GELU
  dim3 gF(DFFN / 128, MROWS / 128);
  enc_gemm_bf16<<<gF, 256, 0, stream>>>(z2h, w1h, b1, nullptr, nullptr, hh, MROWS, DFFN, DD, 2);

  // 8) FFN down + b2 + residual -> final f32 output
  enc_gemm_bf16<<<gD, 256, 0, stream>>>(hh, w2h, b2, z1f, out, nullptr, MROWS, DD, DFFN, 1);
}
